// MoELayer_41188736369136
// MI455X (gfx1250) — matrix operand order repaired and checked by reading, not compiled, not run on hardware
//
#include <hip/hip_runtime.h>
#include <math.h>

#define D_IN   1024
#define D_OUT  1024
#define NEXP   8
#define NTOK   8192
#define TILE_M 64            // tokens per block (4 row-blocks of 16)

typedef __attribute__((ext_vector_type(4)))  float  v4f;
typedef __attribute__((ext_vector_type(8)))  float  v8f;
typedef __attribute__((ext_vector_type(8)))  __bf16 v8bf;
typedef __attribute__((ext_vector_type(16))) __bf16 v16bf;

// ---------------------------------------------------------------------------
// A-fragment (16x32 bf16) from an f32 row pointer (row base + kb already added).
// Layout: lane group g = lane>>4. dwords 0-3 hold K = g*16 + 0..7 (2/dword),
// dwords 4-7 hold K = g*16 + 8..15.
// ---------------------------------------------------------------------------
__device__ __forceinline__ v16bf pack_a_f32(const float* __restrict__ p, int half) {
    v4f a0 = *(const v4f*)(p + half * 16);
    v4f a1 = *(const v4f*)(p + half * 16 + 4);
    v4f a2 = *(const v4f*)(p + half * 16 + 8);
    v4f a3 = *(const v4f*)(p + half * 16 + 12);
    v16bf r;
#pragma unroll
    for (int i = 0; i < 4; ++i) {
        r[i]      = (__bf16)a0[i];
        r[4 + i]  = (__bf16)a1[i];
        r[8 + i]  = (__bf16)a2[i];
        r[12 + i] = (__bf16)a3[i];
    }
    return r;
}

// Same A layout but sourcing bf16 (LDS h tile) -> ds_load_b128 x2.
__device__ __forceinline__ v16bf pack_a_bf16(const __bf16* __restrict__ p, int half) {
    v8bf lo = *(const v8bf*)(p + half * 16);
    v8bf hi = *(const v8bf*)(p + half * 16 + 8);
    v16bf r;
#pragma unroll
    for (int i = 0; i < 8; ++i) { r[i] = lo[i]; r[8 + i] = hi[i]; }
    return r;
}

// ---------------------------------------------------------------------------
// 0) init: zero output + expert counters
// ---------------------------------------------------------------------------
__global__ __launch_bounds__(256) void init_kernel(float* __restrict__ out,
                                                   long long n,
                                                   int* __restrict__ cnt) {
    long long i = (long long)blockIdx.x * 256 + threadIdx.x;
    if (i < n) out[i] = 0.0f;
    if (blockIdx.x == 0 && threadIdx.x < NEXP) cnt[threadIdx.x] = 0;
}

// ---------------------------------------------------------------------------
// 1) W [E][K][N] f32  ->  Wt [E][N][K] bf16  (32x32 LDS tile transpose)
//    grid: x = 1024 tiles, y = expert
// ---------------------------------------------------------------------------
__global__ __launch_bounds__(256) void transpose_w_bf16(const float* __restrict__ W,
                                                        __bf16* __restrict__ Wt) {
    const int e  = blockIdx.y;
    const int t  = blockIdx.x;
    const int kb = (t & 31) * 32;
    const int nb = (t >> 5) * 32;
    __shared__ float tile[32][33];
    const int c  = threadIdx.x & 31;
    const int r0 = threadIdx.x >> 5;  // 0..7
    const float* Wp = W + (size_t)e * D_IN * D_OUT;
#pragma unroll
    for (int i = 0; i < 4; ++i) {
        int r = r0 + i * 8;
        tile[r][c] = Wp[(size_t)(kb + r) * D_OUT + (nb + c)];
    }
    __syncthreads();
    __bf16* Wtp = Wt + (size_t)e * D_IN * D_OUT;
#pragma unroll
    for (int i = 0; i < 4; ++i) {
        int r = r0 + i * 8;
        Wtp[(size_t)(nb + r) * D_IN + (kb + c)] = (__bf16)tile[c][r];
    }
}

// ---------------------------------------------------------------------------
// 2) gating: one wave per token. logits = x@Wg + bg, softmax, top-2,
//    atomic-append (token, gate weight) into per-expert routing lists.
// ---------------------------------------------------------------------------
__global__ __launch_bounds__(256) void gating_kernel(const float* __restrict__ x,
                                                     const float* __restrict__ Wg,
                                                     const float* __restrict__ bg,
                                                     int* __restrict__ cnt,
                                                     int* __restrict__ tok_list,
                                                     float* __restrict__ gate_list) {
    const int wave = threadIdx.x >> 5;
    const int lane = threadIdx.x & 31;
    const int n    = blockIdx.x * 8 + wave;

    float acc[NEXP];
#pragma unroll
    for (int e = 0; e < NEXP; ++e) acc[e] = 0.0f;

    const float* xr = x + (size_t)n * D_IN;
#pragma unroll 4
    for (int i = 0; i < 32; ++i) {
        int d = i * 32 + lane;
        float xv = xr[d];
        const float* wr = Wg + (size_t)d * NEXP;
#pragma unroll
        for (int e = 0; e < NEXP; ++e) acc[e] += xv * wr[e];
    }
#pragma unroll
    for (int off = 16; off >= 1; off >>= 1) {
#pragma unroll
        for (int e = 0; e < NEXP; ++e) acc[e] += __shfl_xor(acc[e], off, 32);
    }
    if (lane == 0) {
        float lg[NEXP];
        float m = -1e30f;
#pragma unroll
        for (int e = 0; e < NEXP; ++e) { lg[e] = acc[e] + bg[e]; m = fmaxf(m, lg[e]); }
        float s = 0.0f;
#pragma unroll
        for (int e = 0; e < NEXP; ++e) { lg[e] = __expf(lg[e] - m); s += lg[e]; }
        float inv = 1.0f / s;
        int i0 = 0;
#pragma unroll
        for (int e = 1; e < NEXP; ++e) if (lg[e] > lg[i0]) i0 = e;
        int i1 = (i0 == 0) ? 1 : 0;
#pragma unroll
        for (int e = 0; e < NEXP; ++e) if (e != i0 && lg[e] > lg[i1]) i1 = e;

        int s0 = atomicAdd(&cnt[i0], 1);
        tok_list[i0 * NTOK + s0]  = n;
        gate_list[i0 * NTOK + s0] = lg[i0] * inv;
        int s1 = atomicAdd(&cnt[i1], 1);
        tok_list[i1 * NTOK + s1]  = n;
        gate_list[i1 * NTOK + s1] = lg[i1] * inv;
    }
}

// ---------------------------------------------------------------------------
// 3) fused two-layer expert GEMM over routed token tiles.
//    grid: x = NTOK/64 tiles, y = expert. 512 threads = 16 waves =
//    4 row-blocks x 4 col-groups; each wave owns a 16x256 output slab
//    (16 WMMA accumulator tiles).
//
//    B k-panels are cooperatively staged in LDS in WMMA fragment layout
//    (Bs, 64 KB), prefetched one k-step ahead through registers so the
//    global loads for panel kb+32 are in flight during all 16 WMMAs of
//    panel kb. The h tile for the whole block lives in LDS (129 KB).
//    Total LDS ~194 KB of the WGP's 320 KB.
// ---------------------------------------------------------------------------
__global__ __launch_bounds__(512) void moe_expert_gemm(const float* __restrict__ x,
                                                       const __bf16* __restrict__ W1t,
                                                       const float* __restrict__ b1,
                                                       const __bf16* __restrict__ W2t,
                                                       const float* __restrict__ b2,
                                                       const int* __restrict__ cnt,
                                                       const int* __restrict__ tok_list,
                                                       const float* __restrict__ gate_list,
                                                       float* __restrict__ out) {
    const int e     = blockIdx.y;
    const int nrows = cnt[e];
    const int row0  = blockIdx.x * TILE_M;
    if (row0 >= nrows) return;

    const int tid  = threadIdx.x;
    const int lane = tid & 31;
    const int l16  = lane & 15;
    const int half = lane >> 4;
    const int rm   = (tid >> 5) & 3;  // row-block (16 rows each)
    const int cg   = tid >> 7;        // col-group (256 cols each); waves of a group are tid-contiguous
    const int gtid = tid & 127;       // thread id within col-group

    __shared__ __align__(16) __bf16 hs[TILE_M][D_OUT + 8];   // 129 KB, conflict-free b128
    __shared__ __align__(16) __bf16 Bs[4][16][32][16];       // 64 KB, fragment layout
    __shared__ int   stok[TILE_M];
    __shared__ float sgw[TILE_M];

    if (tid < TILE_M) {
        int r = row0 + tid;
        if (r < nrows) {
            stok[tid] = tok_list[e * NTOK + r];
            sgw[tid]  = gate_list[e * NTOK + r];
        } else {
            stok[tid] = 0;     // read row 0, contribute 0 (gw = 0)
            sgw[tid]  = 0.0f;
        }
    }
    __syncthreads();

    const int colbase = cg * 256;
    const int rowbase = rm * 16;
    const v8f zero = {0.f, 0.f, 0.f, 0.f, 0.f, 0.f, 0.f, 0.f};

    // Each thread stages 4 fragment slots (slot = j*32 + lane_s) of its
    // col-group's 256-col x 32-k panel.
    int s_j[4], s_lane[4];
    const __bf16* s_src[4];  // per-slot column base + kh*16 (add kb per step)
#pragma unroll
    for (int i = 0; i < 4; ++i) {
        int s    = i * 128 + gtid;
        s_j[i]   = s >> 5;
        s_lane[i] = s & 31;
        // filled per phase below (needs w1e/w2e)
        s_src[i] = nullptr;
    }

    // ---------------- Phase A: h = relu(x @ W1[e] + b1[e]) ----------------
    v8f acc[16];
#pragma unroll
    for (int j = 0; j < 16; ++j) acc[j] = zero;

    const float*  xrow = x + (size_t)stok[rowbase + l16] * D_IN;
    const __bf16* w1e  = W1t + (size_t)e * D_IN * D_OUT;
    const __bf16* w2e  = W2t + (size_t)e * D_IN * D_OUT;

#pragma unroll
    for (int i = 0; i < 4; ++i) {
        int colg = s_j[i] * 16 + (s_lane[i] & 15);
        int kh   = s_lane[i] >> 4;
        s_src[i] = w1e + (size_t)(colbase + colg) * D_IN + kh * 16;
    }

    v16bf stage[4];
#pragma unroll
    for (int i = 0; i < 4; ++i) stage[i] = *(const v16bf*)(s_src[i]);  // kb = 0

    for (int kb = 0; kb < D_IN; kb += 32) {
        __syncthreads();   // prior k-step's compute done; safe to overwrite Bs
#pragma unroll
        for (int i = 0; i < 4; ++i)
            *(v16bf*)&Bs[cg][s_j[i]][s_lane[i]][0] = stage[i];
        if (kb + 32 < D_IN) {
#pragma unroll
            for (int i = 0; i < 4; ++i)
                stage[i] = *(const v16bf*)(s_src[i] + kb + 32);  // in flight during WMMAs
        }
        __syncthreads();   // Bs visible to all waves

        v16bf a = pack_a_f32(xrow + kb, half);
#pragma unroll
        for (int j = 0; j < 16; ++j) {
            v16bf b = *(const v16bf*)&Bs[cg][j][lane][0];
            acc[j] = __builtin_amdgcn_wmma_f32_16x16x32_bf16(
                false, a, false, b, (short)0, acc[j], false, false);
        }
    }
#pragma unroll
    for (int j = 0; j < 16; ++j) {
        int col = colbase + j * 16 + l16;
        float bias = b1[e * D_OUT + col];
#pragma unroll
        for (int r = 0; r < 8; ++r) {
            int row = rowbase + r + half * 8;        // C layout: VGPR r, lane>>4 selects M half
            float v = acc[j][r] + bias;
            v = v > 0.0f ? v : 0.0f;
            hs[row][col] = (__bf16)v;
        }
    }
    __syncthreads();

    // ------------- Phase B: out[tok] += gw * (h @ W2[e] + b2[e]) ----------
#pragma unroll
    for (int j = 0; j < 16; ++j) acc[j] = zero;

#pragma unroll
    for (int i = 0; i < 4; ++i) {
        int colg = s_j[i] * 16 + (s_lane[i] & 15);
        int kh   = s_lane[i] >> 4;
        s_src[i] = w2e + (size_t)(colbase + colg) * D_OUT + kh * 16;
    }
#pragma unroll
    for (int i = 0; i < 4; ++i) stage[i] = *(const v16bf*)(s_src[i]);  // kb = 0

    for (int kb = 0; kb < D_OUT; kb += 32) {
        __syncthreads();
#pragma unroll
        for (int i = 0; i < 4; ++i)
            *(v16bf*)&Bs[cg][s_j[i]][s_lane[i]][0] = stage[i];
        if (kb + 32 < D_OUT) {
#pragma unroll
            for (int i = 0; i < 4; ++i)
                stage[i] = *(const v16bf*)(s_src[i] + kb + 32);
        }
        __syncthreads();

        v16bf a = pack_a_bf16(&hs[rowbase + l16][kb], half);  // ds_load_b128 x2
#pragma unroll
        for (int j = 0; j < 16; ++j) {
            v16bf b = *(const v16bf*)&Bs[cg][j][lane][0];
            acc[j] = __builtin_amdgcn_wmma_f32_16x16x32_bf16(
                false, a, false, b, (short)0, acc[j], false, false);
        }
    }
#pragma unroll
    for (int j = 0; j < 16; ++j) {
        int col = colbase + j * 16 + l16;
        float bias = b2[e * D_OUT + col];
#pragma unroll
        for (int r = 0; r < 8; ++r) {
            int row = rowbase + r + half * 8;
            float v = (acc[j][r] + bias) * sgw[row];
            atomicAdd(&out[(size_t)stok[row] * D_OUT + col], v);
        }
    }
}

// ---------------------------------------------------------------------------
extern "C" void kernel_launch(void* const* d_in, const int* in_sizes, int n_in,
                              void* d_out, int out_size, void* d_ws, size_t ws_size,
                              hipStream_t stream) {
    const float* x  = (const float*)d_in[0];
    const float* W1 = (const float*)d_in[1];
    const float* b1 = (const float*)d_in[2];
    const float* W2 = (const float*)d_in[3];
    const float* b2 = (const float*)d_in[4];
    const float* Wg = (const float*)d_in[5];
    const float* bg = (const float*)d_in[6];
    float* out = (float*)d_out;

    // workspace layout (~34.1 MB)
    unsigned char* ws = (unsigned char*)d_ws;
    size_t off = 0;
    __bf16* W1t = (__bf16*)(ws + off); off += (size_t)NEXP * D_IN * D_OUT * sizeof(__bf16);
    __bf16* W2t = (__bf16*)(ws + off); off += (size_t)NEXP * D_IN * D_OUT * sizeof(__bf16);
    int*   cnt       = (int*)(ws + off);   off += 256;
    int*   tok_list  = (int*)(ws + off);   off += (size_t)NEXP * NTOK * sizeof(int);
    float* gate_list = (float*)(ws + off); off += (size_t)NEXP * NTOK * sizeof(float);
    (void)off; (void)ws_size; (void)in_sizes; (void)n_in; (void)out_size;

    const long long out_elems = (long long)NTOK * D_OUT;
    init_kernel<<<(unsigned)((out_elems + 255) / 256), 256, 0, stream>>>(out, out_elems, cnt);

    transpose_w_bf16<<<dim3(1024, NEXP), 256, 0, stream>>>(W1, W1t);
    transpose_w_bf16<<<dim3(1024, NEXP), 256, 0, stream>>>(W2, W2t);

    gating_kernel<<<NTOK / 8, 256, 0, stream>>>(x, Wg, bg, cnt, tok_list, gate_list);

    moe_expert_gemm<<<dim3(NTOK / TILE_M, NEXP), 512, 0, stream>>>(
        x, W1t, b1, W2t, b2, cnt, tok_list, gate_list, out);
}
